// GegenbauerKANLayer_10617159155738
// MI455X (gfx1250) — compile-verified, matrix operand order checked by reading, not run on hardware
//
#include <hip/hip_runtime.h>

typedef __attribute__((ext_vector_type(16))) _Float16 v16h;
typedef __attribute__((ext_vector_type(8)))  _Float16 v8h;
typedef __attribute__((ext_vector_type(8)))  float    v8f;

#define IN_DIM   256
#define OUT_DIM  256
#define DEG      8
#define NCOEF    (DEG + 1)            // 9
#define KDIM     (IN_DIM * NCOEF)     // 2304, ordered d-major: k = d*256 + i
#define BATCH    65536
#define MT       64                   // batch rows per workgroup (4 M-tiles)
#define NSLAB    (KDIM / 32)          // 72 K-slabs of 32
#define SLAB_H   (MT * 32)            // 2048 f16 per slab (64 rows x 32 k)

// tanh(x) = 1 - 2/(e^{2x} + 1); exact at +/-inf, error far below f16 storage ulp.
__device__ __forceinline__ float fast_tanh(float v) {
    float z = __expf(2.0f * v);                  // v_exp_f32 path
    float r = __builtin_amdgcn_rcpf(z + 1.0f);   // v_rcp_f32
    return fmaf(-2.0f, r, 1.0f);
}

// ---------------------------------------------------------------------------
// Prep: coeffs [i, o, d] (f32) -> WT [o, d*256 + i] (f16)  (d-major K order,
// matching the LDS G layout). 2304*256 elements, one per thread.
// ---------------------------------------------------------------------------
__global__ __launch_bounds__(256)
void gegen_prep_wt(const float* __restrict__ coeffs, _Float16* __restrict__ WT) {
    int idx = blockIdx.x * 256 + threadIdx.x;   // idx = o*KDIM + k
    int o = idx / KDIM;
    int k = idx - o * KDIM;
    int d = k >> 8;                             // k = d*256 + i
    int i = k & 255;
    WT[idx] = (_Float16)coeffs[((size_t)i * OUT_DIM + o) * NCOEF + d];
}

// ---------------------------------------------------------------------------
// Fused: tanh + Gegenbauer recurrence -> swizzled K-blocked f16 LDS tile
// Gs[kb][row][32] (chunk c of row r stored at c ^ ((r>>2)&3); conflict-free
// ds_load_b128 with pure immediate offsets). Then WMMA GEMM against WT:
// 8 waves, each 4 M-tiles x 2 N-tiles (B reused 4x, A reused 2x).
// ---------------------------------------------------------------------------
__global__ __launch_bounds__(256)
void gegen_kan_wmma(const float* __restrict__ x,
                    const _Float16* __restrict__ WT,
                    float* __restrict__ out) {
    __shared__ __attribute__((aligned(16))) _Float16 Gs[NSLAB * SLAB_H]; // 288 KB

    const int tid = threadIdx.x;
    const int b0  = blockIdx.x * MT;

    // ---- Phase 1: Gegenbauer features (thread -> input column i = tid) ----
    // Store target for (row j, degree d): slab kb = d*8 + (i>>5), kk = i&31,
    // swizzled chunk = ((i>>3)&3) ^ ((j>>2)&3).
    const float alpha = 1.0f;
    const int   i5    = tid >> 5;          // slab within each d-group
    const int   cch   = (tid >> 3) & 3;    // pre-swizzle chunk
    const int   i7    = tid & 7;           // position within chunk

    #pragma unroll 4
    for (int j = 0; j < MT; ++j) {
        float t = fast_tanh(x[(size_t)(b0 + j) * IN_DIM + tid]);
        float c[NCOEF];
        c[0] = 1.0f;
        c[1] = 2.0f * alpha * t;
        #pragma unroll
        for (int n = 1; n < DEG; ++n) {
            // all-constant operands -> compile-time folded multipliers (no div)
            const float k1 = 2.0f * ((float)n + alpha) / ((float)n + 1.0f);
            const float k2 = ((float)n + 2.0f * alpha - 1.0f) / ((float)n + 1.0f);
            c[n + 1] = fmaf(k1 * t, c[n], -k2 * c[n - 1]);
        }
        const int cs   = cch ^ ((j >> 2) & 3);
        const int base = i5 * SLAB_H + j * 32 + cs * 8 + i7;
        #pragma unroll
        for (int d = 0; d < NCOEF; ++d)
            Gs[base + d * 8 * SLAB_H] = (_Float16)c[d];
    }
    __syncthreads();

    // ---- Phase 2: WMMA GEMM over K = 2304 ----
    const int lane = tid & 31;
    const int wave = tid >> 5;        // 0..7
    const int half = lane >> 4;       // 0: K {0..7,16..23}; 1: K {8..15,24..31}
    const int lrow = lane & 15;       // A row / B column within a 16x16 tile
    const int n0   = wave * 2;        // this wave's two N-tiles: n0, n0+1

    const _Float16* wt0 = WT + (size_t)(n0 * 16 + lrow) * KDIM;

    // Per-lane swizzled in-slab offsets (halves); per-m offset is m*512 (imm).
    const int sw  = (lrow >> 2) & 3;
    const int oLo = lrow * 32 + ((half ^ sw) * 8);          // K chunk half
    const int oHi = lrow * 32 + (((half + 2) ^ sw) * 8);    // K chunk half+2

    v8f acc[4][2];
    #pragma unroll
    for (int m = 0; m < 4; ++m) {
        acc[m][0] = (v8f){};
        acc[m][1] = (v8f){};
    }

    int sb = 0;                                   // slab base (halves)
    for (int kb = 0; kb < KDIM; kb += 32) {
        const int koff = kb + half * 8;

        // B fragments for the wave's two N-tiles (streamed from L2-resident WT)
        v16h bv[2];
        #pragma unroll
        for (int t = 0; t < 2; ++t) {
            const _Float16* wp = wt0 + (size_t)(t * 16) * KDIM + koff;
            v8h lo = *(const v8h*)(wp);
            v8h hi = *(const v8h*)(wp + 16);
            #pragma unroll
            for (int e = 0; e < 8; ++e) { bv[t][e] = lo[e]; bv[t][e + 8] = hi[e]; }
        }

        // A fragments: immediate-offset ds_load_b128 off two running bases
        #pragma unroll
        for (int m = 0; m < 4; ++m) {
            v8h lo = *(const v8h*)&Gs[sb + m * 512 + oLo];
            v8h hi = *(const v8h*)&Gs[sb + m * 512 + oHi];
            v16h a;
            #pragma unroll
            for (int e = 0; e < 8; ++e) { a[e] = lo[e]; a[e + 8] = hi[e]; }

            acc[m][0] = __builtin_amdgcn_wmma_f32_16x16x32_f16(
                false, a, false, bv[0], (short)0, acc[m][0], false, false);
            acc[m][1] = __builtin_amdgcn_wmma_f32_16x16x32_f16(
                false, a, false, bv[1], (short)0, acc[m][1], false, false);
        }
        sb += SLAB_H;
    }

    // ---- Store D: VGPR v, lane<16 -> (M=v, N=lane); lane>=16 -> (M=v+8, N=lane-16)
    const int mbase = half * 8;
    #pragma unroll
    for (int m = 0; m < 4; ++m) {
        float* o0 = out + (size_t)(b0 + m * 16 + mbase) * OUT_DIM + n0 * 16 + lrow;
        #pragma unroll
        for (int v = 0; v < 8; ++v) {
            o0[(size_t)v * OUT_DIM]      = acc[m][0][v];
            o0[(size_t)v * OUT_DIM + 16] = acc[m][1][v];
        }
    }
}

// ---------------------------------------------------------------------------
extern "C" void kernel_launch(void* const* d_in, const int* in_sizes, int n_in,
                              void* d_out, int out_size, void* d_ws, size_t ws_size,
                              hipStream_t stream) {
    const float* x      = (const float*)d_in[0];                 // [65536, 256] f32
    const float* coeffs = (const float*)d_in[1];                 // [256, 256, 9] f32
    _Float16*    WT     = (_Float16*)d_ws;                       // [256, 2304] f16 (1.15 MB)
    float*       out    = (float*)d_out;                         // [65536, 256] f32

    gegen_prep_wt<<<(KDIM * OUT_DIM) / 256, 256, 0, stream>>>(coeffs, WT);
    gegen_kan_wmma<<<BATCH / MT, 256, 0, stream>>>(x, WT, out);
}
